// lstm_decoder_51625506898451
// MI455X (gfx1250) — matrix operand order repaired and checked by reading, not compiled, not run on hardware
//
#include <hip/hip_runtime.h>
#include <hip/hip_bf16.h>
#include <cstddef>
#include <cstdint>

// ---------------------------------------------------------------------------
// LSTM decoder, persistent-kernel formulation for MI455X (gfx1250, wave32).
//   grid = 32 WGs (one 16-row batch tile each), block = 256 threads (8 waves).
//   Each WG runs all 512 timesteps; c-state lives in registers, h in LDS,
//   weights stream from L2 as bf16 row-major over K (each B lane reads 16 contiguous K).
//   - Weight pointers are laundered (as integers, recast to address_space(1))
//     inside the time loop: LICM cannot hoist the weight loads (previously
//     spilled ~1000 VGPRs to scratch), yet they stay plain global_load_b128.
//   - kc-outer / (p,g)-inner WMMA loop: one A fragment live at a time, eight
//     independent accumulator chains -> low VGPR pressure + deep pipelining.
//   - Accumulators start from inline-zero C; bias folded into the epilogue.
// ---------------------------------------------------------------------------

typedef __attribute__((ext_vector_type(16))) __bf16 v16bf;
typedef __attribute__((ext_vector_type(8)))  __bf16 v8bf;
typedef __attribute__((ext_vector_type(4)))  __bf16 v4bf;
typedef __attribute__((ext_vector_type(8)))  float  v8f;
typedef __attribute__((ext_vector_type(4)))  float  v4f;

typedef __attribute__((address_space(1))) __bf16  gbf16;
typedef __attribute__((address_space(1))) v16bf   gv16bf;
typedef __attribute__((address_space(1))) v4f     gv4f;

static constexpr int BATCH   = 512;
static constexpr int SEQL    = 512;
static constexpr int HID     = 256;   // LATENT
static constexpr int MFEAT   = 17;
static constexpr int INSZ    = MFEAT + HID; // 273
static constexpr int GATES   = 4 * HID;     // 1024
static constexpr int TILE_B  = 16;          // batch rows per WG

// workspace layout (bytes)
static constexpr size_t WS_WHH  = 0;                                   // 1024*256 bf16
static constexpr size_t WS_WX   = WS_WHH + (size_t)GATES * HID * 2;    // 1024*256 bf16
static constexpr size_t WS_WO   = WS_WX  + (size_t)GATES * HID * 2;    // 1024*32  bf16
static constexpr size_t WS_BIAS = WS_WO  + (size_t)GATES * 32  * 2;    // 1024 f32

__device__ __forceinline__ float sgm(float x)  { return 1.0f / (1.0f + __expf(-x)); }
__device__ __forceinline__ float tnh(float x)  { return 1.0f - 2.0f / (__expf(2.0f * x) + 1.0f); }

__device__ __forceinline__ v16bf make_a(const __bf16* p0, const __bf16* p1) {
    // 16-bit A fragment (16x32): per lane two runs of 8 consecutive K values.
    v8bf lo = *(const v8bf*)p0;
    v8bf hi = *(const v8bf*)p1;
    v16bf a;
#pragma unroll
    for (int i = 0; i < 8; ++i) { a[i] = lo[i]; a[i + 8] = hi[i]; }
    return a;
}

__device__ __forceinline__ v8f wmma_bf16(v16bf a, v16bf b, v8f c) {
    return __builtin_amdgcn_wmma_f32_16x16x32_bf16(false, a, false, b, (short)0, c, false, false);
}

// ---------------------------------------------------------------------------
// Prep: fp32 -> bf16 weight conversion (row-major over K; this code's B-fragment
// order: each lane of a B fragment reads 16 contiguous K of one output row),
// split W_ih into z-part (cols 17..272) and out_prev-part (cols 0..16, padded
// to K=32 with zeros), and fold the two bias vectors.
// ---------------------------------------------------------------------------
__global__ void lstm_prep_kernel(const float* __restrict__ W_ih,
                                 const float* __restrict__ W_hh,
                                 const float* __restrict__ b_ih,
                                 const float* __restrict__ b_hh,
                                 __bf16* __restrict__ w_hh_b,
                                 __bf16* __restrict__ w_x_b,
                                 __bf16* __restrict__ w_o_b,
                                 float*  __restrict__ bias) {
    const int tid = blockIdx.x * blockDim.x + threadIdx.x;
    const int nt  = gridDim.x * blockDim.x;
    for (int i = tid; i < GATES * HID; i += nt)
        w_hh_b[i] = (__bf16)W_hh[i];
    for (int i = tid; i < GATES * HID; i += nt) {
        int r = i >> 8, c = i & 255;
        w_x_b[i] = (__bf16)W_ih[r * INSZ + MFEAT + c];
    }
    for (int i = tid; i < GATES * 32; i += nt) {
        int r = i >> 5, c = i & 31;
        w_o_b[i] = (c < MFEAT) ? (__bf16)W_ih[r * INSZ + c] : (__bf16)0.0f;
    }
    for (int i = tid; i < GATES; i += nt)
        bias[i] = b_ih[i] + b_hh[i];
}

// ---------------------------------------------------------------------------
// Main persistent kernel.
// ---------------------------------------------------------------------------
__global__ __launch_bounds__(256) void lstm_main_kernel(
        const float* __restrict__ in_seq,   // (B, L, H)
        const float* __restrict__ h0,       // (1, B, H)
        const float* __restrict__ c0,       // (1, B, H)
        const __bf16* __restrict__ w_hh_b,  // (1024, 256) bf16
        const __bf16* __restrict__ w_x_b,   // (1024, 256) bf16
        const __bf16* __restrict__ w_o_b,   // (1024, 32)  bf16
        const float* __restrict__ bias,     // (1024)
        const float* __restrict__ W_lin,    // (17, 256)
        const float* __restrict__ b_lin,    // (17)
        float* __restrict__ out)            // (B, L, 17)
{
    __shared__ float  lds_h [TILE_B][HID];   // h_t fp32 (for output linear)
    __shared__ __bf16 lds_hb[TILE_B][HID];   // h_t bf16 (WMMA A source)
    __shared__ __bf16 lds_zb[TILE_B][HID];   // z_t bf16
    __shared__ __bf16 lds_xo[TILE_B][32];    // out_prev bf16, K-padded to 32
    __shared__ float  lds_out[TILE_B][MFEAT];

    const int tid        = threadIdx.x;
    const int lane       = tid & 31;
    const int wv         = tid >> 5;             // wave 0..7: hidden slice [wv*32, wv*32+32)
    const int batch_base = blockIdx.x * TILE_B;
    const int mA  = lane & 15;                   // row (A) / N-within-tile (B,C)
    const int k0  = (lane >> 4) << 4;            // A-fragment K sub-offset: 0 or 16
    const int kB  = (lane >> 4) << 4;            // B-fragment K sub-offset: 0 or 16
    const int rC  = (lane >> 4) << 3;            // C-layout row offset: 0 or 8

    // ---- initial state ----
    for (int idx = tid; idx < TILE_B * HID; idx += 256) {
        int r = idx >> 8, c = idx & 255;
        float hv = h0[(size_t)(batch_base + r) * HID + c];
        lds_h[r][c]  = hv;
        lds_hb[r][c] = (__bf16)hv;
    }
    for (int idx = tid; idx < TILE_B * MFEAT; idx += 256)
        lds_out[idx / MFEAT][idx % MFEAT] = 0.0f;   // start token

    float c_reg[2][8];     // cell state, C-matrix layout, in VGPRs for all 512 steps
    float bias_v[2][4];    // folded bias, per (p, gate) -- loop-invariant
#pragma unroll
    for (int p = 0; p < 2; ++p) {
        const int j = wv * 32 + p * 16 + mA;
#pragma unroll
        for (int e = 0; e < 8; ++e)
            c_reg[p][e] = c0[(size_t)(batch_base + e + rC) * HID + j];
#pragma unroll
        for (int g = 0; g < 4; ++g)
            bias_v[p][g] = bias[(g * 16 + 2 * wv + p) * 16 + mA];
    }
    __syncthreads();

#pragma unroll 1
    for (int t = 0; t < SEQL; ++t) {
        // Launder the loop-invariant weight pointers (as integers, recast to
        // address_space(1)) so LICM cannot hoist the per-step weight loads out
        // of the 512-iteration loop, while codegen stays global_load_b128.
        uintptr_t whh_u = (uintptr_t)w_hh_b;
        uintptr_t wx_u  = (uintptr_t)w_x_b;
        uintptr_t wo_u  = (uintptr_t)w_o_b;
        asm volatile("" : "+s"(whh_u), "+s"(wx_u), "+s"(wo_u));
        const gbf16* whh_p = (const gbf16*)whh_u;
        const gbf16* wx_p  = (const gbf16*)wx_u;
        const gbf16* wo_p  = (const gbf16*)wo_u;

        // ---- phase 1: stage z_t and out_prev into LDS as bf16 ----
        {
            const v4f* seq4 = (const v4f*)in_seq;
#pragma unroll
            for (int i = 0; i < 4; ++i) {
                int idx = tid + i * 256;           // 0..1023 float4s
                int r = idx >> 6, c4 = idx & 63;   // 64 float4 per row
                v4f v = seq4[((size_t)(batch_base + r) * SEQL + t) * 64 + c4];
                v4bf b4 = { (__bf16)v.x, (__bf16)v.y, (__bf16)v.z, (__bf16)v.w };
                *(v4bf*)&lds_zb[r][c4 * 4] = b4;
            }
        }
        for (int idx = tid; idx < TILE_B * 32; idx += 256) {
            int r = idx >> 5, c = idx & 31;
            lds_xo[r][c] = (c < MFEAT) ? (__bf16)lds_out[r][c] : (__bf16)0.0f;
        }
        __syncthreads();

        // ---- phase 2: WMMA gates, kc-outer so only one A fragment is live.
        // Unified K index: kc 0..7 -> h (w_hh), 8..15 -> z (w_x), 16 -> out_prev.
        // Wave wv owns hidden cols [wv*32, wv*32+32): tiles n = g*16 + 2*wv + p,
        // so i/f/g/o for the same hidden units land in the same wave.
        v8f acc[2][4];
#pragma unroll
        for (int p = 0; p < 2; ++p)
#pragma unroll
            for (int g = 0; g < 4; ++g)
                acc[p][g] = (v8f){};   // inline-zero C for the first WMMA

#pragma unroll
        for (int kc = 0; kc < 17; ++kc) {
            v16bf a;
            if (kc < 8)
                a = make_a(&lds_hb[mA][kc * 32 + k0],
                           &lds_hb[mA][kc * 32 + k0 + 8]);
            else if (kc < 16)
                a = make_a(&lds_zb[mA][(kc - 8) * 32 + k0],
                           &lds_zb[mA][(kc - 8) * 32 + k0 + 8]);
            else
                a = make_a(&lds_xo[mA][k0], &lds_xo[mA][k0 + 8]);

#pragma unroll
            for (int p = 0; p < 2; ++p) {
#pragma unroll
                for (int g = 0; g < 4; ++g) {
                    const int n = g * 16 + 2 * wv + p;
                    const size_t row = (size_t)(n * 16 + mA);
                    v16bf b;
                    if (kc < 8)
                        b = *(const gv16bf*)(whh_p + row * HID + kc * 32 + kB);
                    else if (kc < 16)
                        b = *(const gv16bf*)(wx_p + row * HID + (kc - 8) * 32 + kB);
                    else
                        b = *(const gv16bf*)(wo_p + row * 32 + kB);
                    acc[p][g] = wmma_bf16(a, b, acc[p][g]);
                }
            }
        }

        // ---- epilogue: LSTM cell, entirely in this wave's registers ----
        float hreg[2][8];
#pragma unroll
        for (int p = 0; p < 2; ++p) {
#pragma unroll
            for (int e = 0; e < 8; ++e) {
                float ig = sgm(acc[p][0][e] + bias_v[p][0]);
                float fg = sgm(acc[p][1][e] + bias_v[p][1]);
                float gg = tnh(acc[p][2][e] + bias_v[p][2]);
                float og = sgm(acc[p][3][e] + bias_v[p][3]);
                float cn = fg * c_reg[p][e] + ig * gg;
                c_reg[p][e] = cn;
                hreg[p][e] = og * tnh(cn);
            }
        }
        __syncthreads();   // all A-fragment reads of lds_hb/zb/xo are done

        // ---- h writeback (C-matrix layout -> LDS, fp32 + bf16) ----
#pragma unroll
        for (int p = 0; p < 2; ++p) {
            const int j = wv * 32 + p * 16 + mA;
#pragma unroll
            for (int e = 0; e < 8; ++e) {
                const int r = e + rC;
                lds_h[r][j]  = hreg[p][e];
                lds_hb[r][j] = (__bf16)hreg[p][e];
            }
        }
        __syncthreads();

        // ---- phase 3: out_t = h_t @ W_lin^T + b_lin  (16x17, tiny -> VALU) ----
        uintptr_t wl_u = (uintptr_t)W_lin;
        asm volatile("" : "+s"(wl_u));     // same anti-hoisting treatment
        const gv4f* wlin4 = (const gv4f*)wl_u;
        for (int idx = tid; idx < TILE_B * MFEAT; idx += 256) {
            const int r = idx / MFEAT, c = idx % MFEAT;
            const v4f* hrow = (const v4f*)&lds_h[r][0];
            float s = b_lin[c];
#pragma unroll 4
            for (int k4 = 0; k4 < HID / 4; ++k4) {
                v4f w4 = wlin4[c * (HID / 4) + k4];
                v4f h4 = hrow[k4];
                s += h4.x * w4.x + h4.y * w4.y + h4.z * w4.z + h4.w * w4.w;
            }
            lds_out[r][c] = s;
            out[((size_t)(batch_base + r) * SEQL + t) * MFEAT + c] = s;
        }
        __syncthreads();
    }
}

extern "C" void kernel_launch(void* const* d_in, const int* in_sizes, int n_in,
                              void* d_out, int out_size, void* d_ws, size_t ws_size,
                              hipStream_t stream) {
    (void)in_sizes; (void)n_in; (void)out_size; (void)ws_size;

    const float* in_seq = (const float*)d_in[0];
    const float* h0     = (const float*)d_in[1];
    const float* c0     = (const float*)d_in[2];
    const float* W_ih   = (const float*)d_in[3];
    const float* W_hh   = (const float*)d_in[4];
    const float* b_ih   = (const float*)d_in[5];
    const float* b_hh   = (const float*)d_in[6];
    const float* W_lin  = (const float*)d_in[7];
    const float* b_lin  = (const float*)d_in[8];
    float* out          = (float*)d_out;

    char* ws = (char*)d_ws;
    __bf16* w_hh_b = (__bf16*)(ws + WS_WHH);
    __bf16* w_x_b  = (__bf16*)(ws + WS_WX);
    __bf16* w_o_b  = (__bf16*)(ws + WS_WO);
    float*  bias   = (float*)(ws + WS_BIAS);

    lstm_prep_kernel<<<256, 256, 0, stream>>>(W_ih, W_hh, b_ih, b_hh,
                                              w_hh_b, w_x_b, w_o_b, bias);

    lstm_main_kernel<<<BATCH / TILE_B, 256, 0, stream>>>(
        in_seq, h0, c0, w_hh_b, w_x_b, w_o_b, bias, W_lin, b_lin, out);
}
